// M_40879498725946
// MI455X (gfx1250) — matrix operand order repaired and checked by reading, not compiled, not run on hardware
//
#include <hip/hip_runtime.h>
#include <hip/hip_bf16.h>

#define BATCH    128
#define CH       672
#define HW       196           // 14*14
#define MT       96            // output-channel rows per workgroup (6 waves x 16)
#define KT       32            // K step = bf16 WMMA K
#define NT       13            // ceil(196/16) 16-wide N tiles
#define NPAD     208           // NT*16, zero-padded columns
#define BSTR     40            // LDS row stride for sB in bf16 elems (80B, 16B-aligned)
#define NTHREADS 192
#define EPS      1e-5f

typedef __bf16 bf16;
typedef __attribute__((ext_vector_type(16))) bf16  v16bf;
typedef __attribute__((ext_vector_type(8)))  bf16  v8bf;
typedef __attribute__((ext_vector_type(8)))  float v8f;

__global__ __launch_bounds__(NTHREADS)
void fused_bn_relu_gemm(const float* __restrict__ x120,
                        const float* __restrict__ x129,
                        const float* __restrict__ gamma,
                        const float* __restrict__ beta,
                        const float* __restrict__ rmean,
                        const float* __restrict__ rvar,
                        const float* __restrict__ w,
                        float* __restrict__ out)
{
    __shared__ float s_scale[CH];
    __shared__ float s_shift[CH];
    __shared__ __align__(16) bf16 sA[MT * KT];      // weights tile [m][k], row-major
    __shared__ __align__(16) bf16 sB[NPAD * BSTR];  // activation tile [n][k] (transposed)

    const int tid   = threadIdx.x;
    const int lane  = tid & 31;
    const int wave  = tid >> 5;
    const int batch = blockIdx.x / (CH / MT);
    const int mBase = (blockIdx.x % (CH / MT)) * MT;

    // Fold BN into per-channel scale/shift once per block (hits L2 across blocks).
    for (int c = tid; c < CH; c += NTHREADS) {
        float s = gamma[c] * rsqrtf(rvar[c] + EPS);
        s_scale[c] = s;
        s_shift[c] = beta[c] - rmean[c] * s;
    }

    v8f acc[NT];
    const v8f vzero = {0.f, 0.f, 0.f, 0.f, 0.f, 0.f, 0.f, 0.f};
#pragma unroll
    for (int t = 0; t < NT; ++t) acc[t] = vzero;

    const int n_frag = lane & 15;
    const int kbaseA = (lane < 16) ? 0 : 16;  // A frag: 16 consecutive K per half-wave
    const int kbaseB = (lane < 16) ? 0 : 16;  // B frag: 16 consecutive K per half-wave
    const size_t xoff = (size_t)batch * CH * HW;

    for (int k0 = 0; k0 < CH; k0 += KT) {
        __syncthreads();  // previous iteration finished reading LDS

        // Stage weights: W[mBase+r][k0+k] -> sA (fp32 -> bf16). 16 elems/thread.
#pragma unroll
        for (int i = 0; i < (MT * KT) / NTHREADS; ++i) {
            int idx = tid + i * NTHREADS;
            int r   = idx >> 5;   // /KT
            int k   = idx & 31;   // %KT
            sA[idx] = (bf16)w[(size_t)(mBase + r) * CH + (k0 + k)];
        }

        // Stage activations fused with BN+add+ReLU, transposed [n][k], zero-padded.
        for (int idx = tid; idx < KT * NPAD; idx += NTHREADS) {
            int c = idx / NPAD;          // 0..31 (const divide -> mul/shift)
            int n = idx - c * NPAD;      // 0..207
            float z = 0.f;
            if (n < HW) {
                int ch   = k0 + c;
                size_t g = xoff + (size_t)ch * HW + n;   // coalesced over n
                z = fmaxf(x120[g] * s_scale[ch] + s_shift[ch] + x129[g], 0.f);
            }
            sB[n * BSTR + c] = (bf16)z;
        }
        __syncthreads();

        // A fragment (16x32 bf16, 16 consecutive K per half-wave): two ds_load_b128 per lane.
        union { v16bf v; v8bf h[2]; } afr;
        const bf16* arow = &sA[(wave * 16 + (lane & 15)) * KT];
        afr.h[0] = *(const v8bf*)(arow + kbaseA);
        afr.h[1] = *(const v8bf*)(arow + kbaseA + 8);

#pragma unroll
        for (int t = 0; t < NT; ++t) {
            // B fragment: lane = column n, 16 consecutive K, contiguous in LDS.
            v16bf bfr = *(const v16bf*)(&sB[(t * 16 + n_frag) * BSTR + kbaseB]);
            acc[t] = __builtin_amdgcn_wmma_f32_16x16x32_bf16(
                false, afr.v, false, bfr, (short)0, acc[t], false, false);
        }
    }

    // Store per C/D layout: VGPR r -> M = r + (lane<16 ? 0 : 8), N = lane&15.
    const int rowBase = mBase + wave * 16 + ((lane < 16) ? 0 : 8);
#pragma unroll
    for (int t = 0; t < NT; ++t) {
        int n = t * 16 + n_frag;
        if (n < HW) {
            float* op = out + ((size_t)batch * CH + rowBase) * HW + n;
#pragma unroll
            for (int r = 0; r < 8; ++r) op[(size_t)r * HW] = acc[t][r];
        }
    }
}

extern "C" void kernel_launch(void* const* d_in, const int* in_sizes, int n_in,
                              void* d_out, int out_size, void* d_ws, size_t ws_size,
                              hipStream_t stream) {
    (void)in_sizes; (void)n_in; (void)d_ws; (void)ws_size; (void)out_size;
    const float* x120  = (const float*)d_in[0];
    const float* x129  = (const float*)d_in[1];
    const float* gamma = (const float*)d_in[2];
    const float* beta  = (const float*)d_in[3];
    const float* rmean = (const float*)d_in[4];
    const float* rvar  = (const float*)d_in[5];
    const float* convw = (const float*)d_in[6];
    float* out = (float*)d_out;

    dim3 grid(BATCH * (CH / MT));   // 128 * 7 = 896 workgroups
    dim3 block(NTHREADS);           // 6 wave32 waves
    fused_bn_relu_gemm<<<grid, block, 0, stream>>>(
        x120, x129, gamma, beta, rmean, rvar, convw, out);
}
